// MultiHeadAttention_3238405342010
// MI455X (gfx1250) — matrix operand order repaired and checked by reading, compiled, not run on hardware
//
#include <hip/hip_runtime.h>

// ---------------------------------------------------------------------------
// Problem constants (match reference)
// ---------------------------------------------------------------------------
#define NNODES 2048
#define HIDDIM 512
#define NHEADS 8
#define DHEAD  64
#define NEDGES 65536
#define SCALE_QK 0.125f   // 64^-0.5

typedef unsigned short u16;
typedef __attribute__((ext_vector_type(16))) __bf16 v16bf;
typedef __attribute__((ext_vector_type(8)))  float  v8f;

union FragU { v16bf v; unsigned int u[8]; uint4 q[2]; };

// ---------------------------------------------------------------------------
// bf16 <-> f32 helpers (RNE)
// ---------------------------------------------------------------------------
__device__ __forceinline__ u16 f2bf(float f) {
    union { float f; unsigned u; } c; c.f = f;
    unsigned u = c.u + 0x7FFFu + ((c.u >> 16) & 1u);
    return (u16)(u >> 16);
}
__device__ __forceinline__ float bf2f(u16 s) {
    union { unsigned u; float f; } c; c.u = ((unsigned)s) << 16;
    return c.f;
}

// ---------------------------------------------------------------------------
// WMMA wrapper: D = A(16x32 bf16) * B(32x16 bf16) + C(f32)
// ---------------------------------------------------------------------------
__device__ __forceinline__ v8f wmma_bf16(v16bf a, v16bf b, v8f c) {
    return __builtin_amdgcn_wmma_f32_16x16x32_bf16(
        false, a, false, b, (short)0, c, false, false);
}

// A fragment (16x32 bf16) per ISA layout, 2 x ds_load_b128:
//   lanes 0-15: M=lane, K in {0..7, 16..23}; lanes 16-31: M=lane-16, K in {8..15, 24..31}
__device__ __forceinline__ v16bf load_a_frag(const u16* As, int lda, int arow0) {
    int lane = threadIdx.x & 31;
    const u16* p = As + (arow0 + (lane & 15)) * lda + ((lane >> 4) << 3);
    FragU f;
    f.q[0] = *(const uint4*)p;          // K = hs .. hs+7
    f.q[1] = *(const uint4*)(p + 16);   // K = 16+hs .. 16+hs+7
    return f.v;
}

// B fragment (32x16 bf16) from a TRANSPOSED LDS tile BsT[col][k], 2 x ds_load_b128:
//   lanes 0-15: N=lane, K in {0..7, 16..23}; lanes 16-31: N=lane-16, K in {8..15, 24..31}
__device__ __forceinline__ v16bf load_bT_frag(const u16* BsT, int ldk, int bcol0) {
    int lane = threadIdx.x & 31;
    const u16* p = BsT + (bcol0 + (lane & 15)) * ldk + ((lane >> 4) << 3);
    FragU f;
    f.q[0] = *(const uint4*)p;          // K = hs .. hs+7
    f.q[1] = *(const uint4*)(p + 16);   // K = 16+hs .. 16+hs+7
    return f.v;
}

// Cooperative global(bf16)->LDS tile copy using CDNA5 async-to-LDS (ASYNCcnt).
// Bypasses VGPRs entirely; completion via s_wait_asynccnt + workgroup barrier.
template<int ROWS, int COLS, int NT>
__device__ __forceinline__ void load_tile_async(u16* s, const u16* g, int ldg) {
    constexpr int CH  = (ROWS * COLS) / 8;   // 16B chunks
    constexpr int CPR = COLS / 8;            // chunks per row
    static_assert(CH % NT == 0, "tile/threads mismatch");
    int t = threadIdx.x;
#pragma unroll
    for (int i = 0; i < CH; i += NT) {
        int c = i + t;
        int row = c / CPR, off = (c % CPR) * 8;
        unsigned lds = (unsigned)(size_t)(s + row * COLS + off);
        const void* ga = (const void*)(g + (size_t)row * ldg + off);
        asm volatile("global_load_async_to_lds_b128 %0, %1, off"
                     :: "v"(lds), "v"(ga) : "memory");
    }
}
__device__ __forceinline__ void async_wait0() {
    asm volatile("s_wait_asynccnt 0" ::: "memory");
}

// ---------------------------------------------------------------------------
// f32 -> bf16 convert (plain)
// ---------------------------------------------------------------------------
__global__ void k_cvt(const float* __restrict__ src, u16* __restrict__ dst, int n) {
    int i = blockIdx.x * blockDim.x + threadIdx.x;
    if (i < n) dst[i] = f2bf(src[i]);
}

// f32 [512][512] -> bf16 transposed [col][k] (for GEMM B operands)
__global__ __launch_bounds__(256) void k_cvt_t(const float* __restrict__ src,
                                               u16* __restrict__ dst) {
    int k = blockIdx.x * 32 + (threadIdx.x & 31);
    int c = blockIdx.y * 8 + (threadIdx.x >> 5);
    dst[(size_t)c * HIDDIM + k] = f2bf(src[(size_t)k * HIDDIM + c]);
}

// ---------------------------------------------------------------------------
// Projection GEMM: C[M x 512] = A[M x 512]bf16 * Wt^T + bias  (Wt is [col][k])
// mode 0: bf16 out [row][512]
// mode 1: bf16 out transposed per-head [H][D][N]  (for V^T)
// mode 2: f32  out [row][512]                     (final output)
// Block = 128 threads (4 waves); block tile 64x64; wave tile 16x64.
// Double-buffered async-to-LDS pipeline over K.
// ---------------------------------------------------------------------------
__global__ __launch_bounds__(128) void k_proj_gemm(
    const u16* __restrict__ A, const u16* __restrict__ Wt,
    const float* __restrict__ bias, void* __restrict__ out, int mode)
{
    __shared__ u16 As[2][64 * 32];
    __shared__ u16 BsT[2][64 * 32];    // [col][k]
    int colTile = blockIdx.x * 64;
    int rowTile = blockIdx.y * 64;
    int wave = threadIdx.x >> 5;
    const u16* Ag = A + (size_t)rowTile * HIDDIM;
    const u16* Bg = Wt + (size_t)colTile * HIDDIM;
    v8f acc[4] = {};

    // prologue: stage tile 0
    load_tile_async<64, 32, 128>(As[0], Ag, HIDDIM);
    load_tile_async<64, 32, 128>(BsT[0], Bg, HIDDIM);

    for (int k0 = 0; k0 < HIDDIM; k0 += 32) {
        int cur = (k0 >> 5) & 1;
        async_wait0();          // this wave's chunks landed
        __syncthreads();        // everyone's chunks landed; prev buffer free
        if (k0 + 32 < HIDDIM) {
            load_tile_async<64, 32, 128>(As[cur ^ 1], Ag + k0 + 32, HIDDIM);
            load_tile_async<64, 32, 128>(BsT[cur ^ 1], Bg + k0 + 32, HIDDIM);
        }
        v16bf af = load_a_frag(As[cur], 32, wave * 16);
#pragma unroll
        for (int j = 0; j < 4; ++j) {
            v16bf bf = load_bT_frag(BsT[cur], 32, j * 16);
            acc[j] = wmma_bf16(af, bf, acc[j]);
        }
    }

    int lane = threadIdx.x & 31;
    int cn = lane & 15;
    int rbase = rowTile + wave * 16 + ((lane >> 4) << 3);
#pragma unroll
    for (int j = 0; j < 4; ++j) {
        int colg = colTile + j * 16 + cn;
        float bv = bias[colg];
#pragma unroll
        for (int r = 0; r < 8; ++r) {
            float v = acc[j][r] + bv;
            int row = rbase + r;
            if (mode == 0) {
                ((u16*)out)[(size_t)row * HIDDIM + colg] = f2bf(v);
            } else if (mode == 1) {
                int hh = colg >> 6, d = colg & 63;
                ((u16*)out)[((size_t)hh * DHEAD + d) * NNODES + row] = f2bf(v);
            } else {
                ((float*)out)[(size_t)row * HIDDIM + colg] = v;
            }
        }
    }
}

// ---------------------------------------------------------------------------
// Scores: S[h][n][m] = SCALE * sum_d Q[n][h*64+d] * K[m][h*64+d], bf16 out
// B-transposed tile = K rows directly ([N][HID] layout). K staged fully (64),
// via async-to-LDS.
// ---------------------------------------------------------------------------
__global__ __launch_bounds__(128) void k_scores(
    const u16* __restrict__ Qb, const u16* __restrict__ Kb, u16* __restrict__ Sb)
{
    __shared__ u16 As[64 * 64];      // [n][d]
    __shared__ u16 BsT[64 * 64];     // [m][d]
    int h = blockIdx.z;
    int colTile = blockIdx.x * 64;   // m
    int rowTile = blockIdx.y * 64;   // n
    int wave = threadIdx.x >> 5;

    load_tile_async<64, 64, 128>(As, Qb + (size_t)rowTile * HIDDIM + h * DHEAD, HIDDIM);
    load_tile_async<64, 64, 128>(BsT, Kb + (size_t)colTile * HIDDIM + h * DHEAD, HIDDIM);
    async_wait0();
    __syncthreads();

    v8f acc[4] = {};
#pragma unroll
    for (int k0 = 0; k0 < 64; k0 += 32) {
        v16bf af = load_a_frag(As + k0, 64, wave * 16);
#pragma unroll
        for (int j = 0; j < 4; ++j) {
            v16bf bf = load_bT_frag(BsT + k0, 64, j * 16);
            acc[j] = wmma_bf16(af, bf, acc[j]);
        }
    }

    int lane = threadIdx.x & 31;
    int cn = lane & 15;
    int rbase = rowTile + wave * 16 + ((lane >> 4) << 3);
    u16* Srow = Sb + (size_t)h * NNODES * NNODES;
#pragma unroll
    for (int j = 0; j < 4; ++j) {
        int colg = colTile + j * 16 + cn;
#pragma unroll
        for (int r = 0; r < 8; ++r)
            Srow[(size_t)(rbase + r) * NNODES + colg] = f2bf(acc[j][r] * SCALE_QK);
    }
}

// ---------------------------------------------------------------------------
// Edge bias: bias[e][h] = edge_attr[e] . We[:,h] + be[h], RMW into Sb
// ---------------------------------------------------------------------------
__global__ __launch_bounds__(256) void k_edge_bias(
    const int* __restrict__ edge_index, const float* __restrict__ edge_attr,
    const float* __restrict__ We, const float* __restrict__ be,
    u16* __restrict__ Sb)
{
    __shared__ float WeL[HIDDIM * NHEADS];   // 16 KB
    for (int i = threadIdx.x; i < HIDDIM * NHEADS; i += 256) WeL[i] = We[i];
    __syncthreads();

    int wave = threadIdx.x >> 5, lane = threadIdx.x & 31;
    int ebase = (blockIdx.x * 8 + wave) * 8;

    for (int ei = 0; ei < 8; ++ei) {
        int e = ebase + ei;
        float acc[8] = {};
        const float* ea = edge_attr + (size_t)e * HIDDIM + lane * 16;
#pragma unroll
        for (int i = 0; i < 16; ++i) {
            float x = ea[i];
            int c = lane * 16 + i;
#pragma unroll
            for (int hh = 0; hh < 8; ++hh) acc[hh] += x * WeL[c * 8 + hh];
        }
#pragma unroll
        for (int off = 16; off > 0; off >>= 1) {
#pragma unroll
            for (int hh = 0; hh < 8; ++hh)
                acc[hh] += __shfl_xor(acc[hh], off, 32);
        }
        if (lane < 8) {
            int row = edge_index[e];
            int col = edge_index[NEDGES + e];
            size_t idx = ((size_t)lane * NNODES + row) * NNODES + col;
            Sb[idx] = f2bf(bf2f(Sb[idx]) + acc[lane] + be[lane]);
        }
    }
}

// ---------------------------------------------------------------------------
// Softmax over key dim: one block (256 thr) per (n,h) row of 2048, in place bf16
// ---------------------------------------------------------------------------
__global__ __launch_bounds__(256) void k_softmax(u16* __restrict__ Sb)
{
    __shared__ float red[256];
    int n = blockIdx.x, h = blockIdx.y, t = threadIdx.x;
    u16* row = Sb + ((size_t)h * NNODES + n) * NNODES;

    float x[8];
#pragma unroll
    for (int i = 0; i < 8; ++i) x[i] = bf2f(row[t * 8 + i]);

    float m = -3.4e38f;
#pragma unroll
    for (int i = 0; i < 8; ++i) m = fmaxf(m, x[i]);
    red[t] = m; __syncthreads();
    for (int off = 128; off > 0; off >>= 1) {
        if (t < off) red[t] = fmaxf(red[t], red[t + off]);
        __syncthreads();
    }
    m = red[0]; __syncthreads();

    float s = 0.f;
#pragma unroll
    for (int i = 0; i < 8; ++i) { x[i] = __expf(x[i] - m); s += x[i]; }
    red[t] = s; __syncthreads();
    for (int off = 128; off > 0; off >>= 1) {
        if (t < off) red[t] += red[t + off];
        __syncthreads();
    }
    float inv = 1.0f / red[0];
#pragma unroll
    for (int i = 0; i < 8; ++i) row[t * 8 + i] = f2bf(x[i] * inv);
}

// ---------------------------------------------------------------------------
// attn @ V: AV[n][h*64+d] = sum_m P[h][n][m] * Vt[h][d][m], K=N=2048
// Double-buffered async-to-LDS pipeline over the key dimension.
// ---------------------------------------------------------------------------
__global__ __launch_bounds__(128) void k_av(
    const u16* __restrict__ Pb, const u16* __restrict__ Vt, u16* __restrict__ AVb)
{
    __shared__ u16 As[2][64 * 32];   // [n][m-chunk]
    __shared__ u16 BsT[2][64 * 32];  // [d][m-chunk]
    int h = blockIdx.z;
    int rowTile = blockIdx.y * 64;                       // n tile; cols = full D
    int wave = threadIdx.x >> 5;
    const u16* Ah = Pb + (size_t)h * NNODES * NNODES + (size_t)rowTile * NNODES;
    const u16* Bh = Vt + (size_t)h * DHEAD * NNODES;     // [d][m], ld = N

    v8f acc[4] = {};
    load_tile_async<64, 32, 128>(As[0], Ah, NNODES);
    load_tile_async<64, 32, 128>(BsT[0], Bh, NNODES);

    for (int k0 = 0; k0 < NNODES; k0 += 32) {
        int cur = (k0 >> 5) & 1;
        async_wait0();
        __syncthreads();
        if (k0 + 32 < NNODES) {
            load_tile_async<64, 32, 128>(As[cur ^ 1], Ah + k0 + 32, NNODES);
            load_tile_async<64, 32, 128>(BsT[cur ^ 1], Bh + k0 + 32, NNODES);
        }
        v16bf af = load_a_frag(As[cur], 32, wave * 16);
#pragma unroll
        for (int j = 0; j < 4; ++j) {
            v16bf bf = load_bT_frag(BsT[cur], 32, j * 16);
            acc[j] = wmma_bf16(af, bf, acc[j]);
        }
    }

    int lane = threadIdx.x & 31;
    int cn = lane & 15;
    int rbase = rowTile + wave * 16 + ((lane >> 4) << 3);
#pragma unroll
    for (int j = 0; j < 4; ++j) {
        int d = j * 16 + cn;
#pragma unroll
        for (int r = 0; r < 8; ++r)
            AVb[(size_t)(rbase + r) * HIDDIM + h * DHEAD + d] = f2bf(acc[j][r]);
    }
}

// ---------------------------------------------------------------------------
// Host-side launcher
// ---------------------------------------------------------------------------
extern "C" void kernel_launch(void* const* d_in, const int* in_sizes, int n_in,
                              void* d_out, int out_size, void* d_ws, size_t ws_size,
                              hipStream_t stream)
{
    (void)in_sizes; (void)n_in; (void)out_size; (void)ws_size;

    const float* h_in = (const float*)d_in[0];
    const int*   ei   = (const int*)d_in[1];
    const float* ea   = (const float*)d_in[2];
    const float* Wq   = (const float*)d_in[3];
    const float* bq   = (const float*)d_in[4];
    const float* Wk   = (const float*)d_in[5];
    const float* bk   = (const float*)d_in[6];
    const float* Wv   = (const float*)d_in[7];
    const float* bv   = (const float*)d_in[8];
    const float* Wo   = (const float*)d_in[9];
    const float* bo   = (const float*)d_in[10];
    const float* We   = (const float*)d_in[11];
    const float* be   = (const float*)d_in[12];

    char* ws = (char*)d_ws;
    size_t cur = 0;
    auto alloc = [&](size_t bytes) -> void* {
        void* p = ws + cur;
        cur = (cur + bytes + 255) & ~(size_t)255;
        return p;
    };
    const size_t NH = (size_t)NNODES * HIDDIM;
    u16* hb   = (u16*)alloc(NH * 2);
    u16* Wqt  = (u16*)alloc((size_t)HIDDIM * HIDDIM * 2);   // transposed [c][k]
    u16* Wkt  = (u16*)alloc((size_t)HIDDIM * HIDDIM * 2);
    u16* Wvt  = (u16*)alloc((size_t)HIDDIM * HIDDIM * 2);
    u16* Wot  = (u16*)alloc((size_t)HIDDIM * HIDDIM * 2);
    u16* Qb   = (u16*)alloc(NH * 2);                        // [N][HID]
    u16* Kb   = (u16*)alloc(NH * 2);                        // [N][HID]
    u16* Vt   = (u16*)alloc((size_t)NHEADS * DHEAD * NNODES * 2);  // [H][D][N]
    u16* AVb  = (u16*)alloc(NH * 2);
    u16* Sb   = (u16*)alloc((size_t)NHEADS * NNODES * NNODES * 2); // scores / probs

    // 1) converts to bf16 (weights transposed for GEMM B operands)
    k_cvt<<<(int)((NH + 255) / 256), 256, 0, stream>>>(h_in, hb, (int)NH);
    dim3 gT(HIDDIM / 32, HIDDIM / 8);
    k_cvt_t<<<gT, 256, 0, stream>>>(Wq, Wqt);
    k_cvt_t<<<gT, 256, 0, stream>>>(Wk, Wkt);
    k_cvt_t<<<gT, 256, 0, stream>>>(Wv, Wvt);
    k_cvt_t<<<gT, 256, 0, stream>>>(Wo, Wot);

    // 2) Q/K/V projections (V stored transposed per head)
    dim3 gP(HIDDIM / 64, NNODES / 64);
    k_proj_gemm<<<gP, 128, 0, stream>>>(hb, Wqt, bq, Qb, 0);
    k_proj_gemm<<<gP, 128, 0, stream>>>(hb, Wkt, bk, Kb, 0);
    k_proj_gemm<<<gP, 128, 0, stream>>>(hb, Wvt, bv, Vt, 1);

    // 3) scores = scale * Q K^T (per head), bf16
    k_scores<<<dim3(NNODES / 64, NNODES / 64, NHEADS), 128, 0, stream>>>(Qb, Kb, Sb);

    // 4) edge bias scatter-add
    k_edge_bias<<<NEDGES / 64, 256, 0, stream>>>(ei, ea, We, be, Sb);

    // 5) softmax over key dim (in place, bf16 probs)
    k_softmax<<<dim3(NNODES, NHEADS), 256, 0, stream>>>(Sb);

    // 6) attn @ V (double-buffered async-to-LDS)
    k_av<<<dim3(1, NNODES / 64, NHEADS), 128, 0, stream>>>(Sb, Vt, AVb);

    // 7) output projection -> f32 d_out
    k_proj_gemm<<<gP, 128, 0, stream>>>(AVb, Wot, bo, (float*)d_out, 2);
}
